// LSTMModel_84696755077531
// MI455X (gfx1250) — compile-verified, matrix operand order checked by reading, not run on hardware
//
#include <hip/hip_runtime.h>

// Problem dims (fixed by reference)
#define Bsz 64
#define Tt  1024
#define Ii  128
#define Hh  512
#define Oo  128
#define G4  2048          // 4*H gate width
#define NWG 32            // workgroups in the persistent scan grid
#define LDPAD 16          // bf16 elements of row padding (32B) -> bank stagger

typedef __attribute__((ext_vector_type(16))) __bf16 bf16x16;
typedef __attribute__((ext_vector_type(8)))  float  v8f;
typedef __attribute__((ext_vector_type(4)))  int    v4i;

typedef __attribute__((address_space(3))) unsigned short lds_us;
typedef __attribute__((address_space(3))) float          lds_f;
typedef __attribute__((address_space(3))) v4i            lds_v4;
typedef __attribute__((address_space(3))) const v4i      lds_cv4;
typedef __attribute__((address_space(1))) v4i            glb_v4;

union Frag { bf16x16 v; v4i q[2]; };

#if __has_builtin(__builtin_amdgcn_global_load_async_to_lds_b128)
#define HAVE_ASYNC_LDS 1
#else
#define HAVE_ASYNC_LDS 0
#endif

// 16-byte global -> LDS copy; async (ASYNCcnt) when available.
__device__ inline void cp16_g2l(const unsigned short* g, lds_us* l) {
#if HAVE_ASYNC_LDS
  __builtin_amdgcn_global_load_async_to_lds_b128((glb_v4*)g, (lds_v4*)l, 0, 0);
#else
  *(lds_v4*)l = *(const v4i*)g;
#endif
}

__device__ inline void async_fence() {
#if HAVE_ASYNC_LDS
#if __has_builtin(__builtin_amdgcn_s_wait_asynccnt)
  __builtin_amdgcn_s_wait_asynccnt(0);
#else
  asm volatile("s_wait_asynccnt 0" ::: "memory");
#endif
#endif
}

// ---------------------------------------------------------------------------
// WMMA 16x16x32 bf16 fragment loaders (ISA §7.12.2, 16-bit A/B layout):
// lane L holds row (row0 + L%16); VGPR0..3 = K base..+7, VGPR4..7 = +16..+23;
// lanes 16..31 use K offset +8  => two contiguous 16-byte loads per lane.
// ---------------------------------------------------------------------------
__device__ inline bf16x16 gload_frag(const unsigned short* __restrict__ base,
                                     int ld, int row0, int k0, int lane) {
  const unsigned short* p =
      base + (size_t)(row0 + (lane & 15)) * ld + k0 + ((lane >> 4) << 3);
  Frag u;
  u.q[0] = *(const v4i*)(p);
  u.q[1] = *(const v4i*)(p + 16);
  return u.v;
}

__device__ inline bf16x16 lds_load_frag(const lds_us* base, int ld, int row0,
                                        int k0, int lane) {
  const lds_us* p = base + (row0 + (lane & 15)) * ld + k0 + ((lane >> 4) << 3);
  Frag u;
  u.q[0] = *(lds_cv4*)(p);
  u.q[1] = *(lds_cv4*)(p + 16);
  return u.v;
}

__device__ inline v8f wmma_bf16(bf16x16 a, bf16x16 b, v8f c) {
  // (neg_a, A, neg_b, B, c_mod, C, reuse_a, reuse_b)
  return __builtin_amdgcn_wmma_f32_16x16x32_bf16(false, a, false, b,
                                                 (short)0, c, false, false);
}

__device__ inline unsigned short f2bf(float f) {  // round-to-nearest-even
  union { float f; unsigned int u; } v; v.f = f;
  unsigned int r = v.u + 0x7FFFu + ((v.u >> 16) & 1u);
  return (unsigned short)(r >> 16);
}

__device__ inline float sigm(float x) { return 1.f / (1.f + __expf(-x)); }

// Device-scope grid barrier (all NWG workgroups co-resident; generation-based,
// wraparound-safe across graph replays).
__device__ inline void grid_barrier(int* cnt, int* gen, int nwg) {
  __threadfence();
  __syncthreads();
  if (threadIdx.x == 0) {
    int g = __hip_atomic_load(gen, __ATOMIC_RELAXED, __HIP_MEMORY_SCOPE_AGENT);
    int v = __hip_atomic_fetch_add(cnt, 1, __ATOMIC_ACQ_REL, __HIP_MEMORY_SCOPE_AGENT);
    if (v == nwg - 1) {
      __hip_atomic_store(cnt, 0, __ATOMIC_RELAXED, __HIP_MEMORY_SCOPE_AGENT);
      __hip_atomic_store(gen, g + 1, __ATOMIC_RELEASE, __HIP_MEMORY_SCOPE_AGENT);
    } else {
      while (__hip_atomic_load(gen, __ATOMIC_ACQUIRE, __HIP_MEMORY_SCOPE_AGENT) == g)
        __builtin_amdgcn_s_sleep(1);
    }
  }
  __syncthreads();
}

// ---------------------------------------------------------------------------
// Prep kernels
// ---------------------------------------------------------------------------
__global__ void init_barrier_kernel(int* p) {
  if (threadIdx.x < 2) p[threadIdx.x] = 0;
}

__global__ void cvt_bf16_kernel(const float* __restrict__ in,
                                unsigned short* __restrict__ out, int n) {
  for (int i = blockIdx.x * blockDim.x + threadIdx.x; i < n;
       i += gridDim.x * blockDim.x)
    out[i] = f2bf(in[i]);
}

__global__ void fuse_bias_kernel(const float* __restrict__ a,
                                 const float* __restrict__ b,
                                 float* __restrict__ o, int n) {
  int i = blockIdx.x * blockDim.x + threadIdx.x;
  if (i < n) o[i] = a[i] + b[i];
}

// ---------------------------------------------------------------------------
// Persistent LSTM layer scan.
//   grid = NWG(32) blocks x 256 threads (8 waves).  WG j owns hidden columns
//   [j*16, j*16+16).  Wave w: gate g=w>>1 (i,f,g,o), M-tiles {2*(w&1), +1}.
//   Per step: stage A-panel [64 x (KX+512)] bf16 (x_t || h_{t-1}) into LDS
//   (async global->LDS when available), gates via WMMA (A from LDS, B=weights
//   streamed from L2 with prefetch), gates -> LDS -> cell update (c persistent
//   in regs) -> bf16 h.  xseq element (b,t,k) at b*x_ld + t*x_ts + k.
// ---------------------------------------------------------------------------
template <int KX>
__global__ __launch_bounds__(256, 1)
void lstm_scan_kernel(const unsigned short* __restrict__ xseq, int x_ld,
                      int x_ts,
                      const unsigned short* __restrict__ Wx,   // [2048, KX]
                      const unsigned short* __restrict__ Wh,   // [2048, 512]
                      const float* __restrict__ bias,          // [2048]
                      unsigned short* __restrict__ hping,      // [2][64][512]
                      unsigned short* __restrict__ hout,       // [B*T, 512]
                      int* cnt, int* gen) {
  constexpr int LDP = KX + Hh + LDPAD;   // padded panel row stride (bf16)
  const int j    = blockIdx.x;           // column-slice owner
  const int tid  = threadIdx.x;
  const int lane = tid & 31;
  const int wave = tid >> 5;
  const int g    = wave >> 1;            // gate index 0..3
  const int mtb  = (wave & 1) * 2;       // M-tile base (0 or 2)
  const int n0   = g * Hh + j * 16;      // weight row base for B fragments

  extern __shared__ char smem_raw[];
  lds_us* panel = (lds_us*)smem_raw;                       // [64][LDP]
  lds_f*  gbuf  = (lds_f*)(smem_raw + (size_t)Bsz * LDP * 2);  // [4][64][16]

  // Persistent cell state + cached biases: thread owns 4 cells,
  // linear p = tid*4+e over [64 rows x 16 cols].
  float cst[4], bI[4], bF[4], bG[4], bO[4];
#pragma unroll
  for (int e = 0; e < 4; ++e) {
    int col = j * 16 + ((tid * 4 + e) & 15);
    cst[e] = 0.f;
    bI[e] = bias[0 * Hh + col];
    bF[e] = bias[1 * Hh + col];
    bG[e] = bias[2 * Hh + col];
    bO[e] = bias[3 * Hh + col];
  }

  // Zero h(t=0) slice owned by this WG.
#pragma unroll
  for (int e = 0; e < 4; ++e) {
    int p = tid * 4 + e, m = p >> 4, n = p & 15;
    hping[m * Hh + j * 16 + n] = 0;
  }
  grid_barrier(cnt, gen, NWG);

  for (int t = 0; t < Tt; ++t) {
    const unsigned short* hprev = hping + ((t & 1) ? (Bsz * Hh) : 0);
    unsigned short* hnext       = hping + ((t & 1) ? 0 : (Bsz * Hh));

    // ---- stage A-panel: rows m: [ x_t(m,0..KX) | h_prev(m,0..511) ] ----
    const unsigned short* xt = xseq + (size_t)t * x_ts;
    constexpr int XQ = KX >> 3;                 // 16B chunks per x row
    for (int i = tid; i < Bsz * XQ; i += 256) {
      int m = i / XQ, c = i - m * XQ;
      cp16_g2l(xt + (size_t)m * x_ld + (c << 3), panel + m * LDP + (c << 3));
    }
    for (int i = tid; i < Bsz * (Hh >> 3); i += 256) {
      int m = i >> 6, c = i & 63;
      cp16_g2l(hprev + m * Hh + (c << 3), panel + m * LDP + KX + (c << 3));
    }
    async_fence();
    __syncthreads();

    v8f acc0 = {}; v8f acc1 = {};

    // --- input contribution: gates += x_t @ Wx^T ---
    for (int k0 = 0; k0 < KX; k0 += 32) {
      bf16x16 bf = gload_frag(Wx, KX, n0, k0, lane);
      if (k0 + 32 < KX)
        __builtin_prefetch(Wx + (size_t)(n0 + (lane & 15)) * KX + k0 + 32, 0, 3);
      bf16x16 a0 = lds_load_frag(panel, LDP, mtb * 16, k0, lane);
      acc0 = wmma_bf16(a0, bf, acc0);
      bf16x16 a1 = lds_load_frag(panel, LDP, (mtb + 1) * 16, k0, lane);
      acc1 = wmma_bf16(a1, bf, acc1);
    }
    // --- recurrent contribution: gates += h_{t-1} @ Wh^T ---
    for (int k0 = 0; k0 < Hh; k0 += 32) {
      bf16x16 bf = gload_frag(Wh, Hh, n0, k0, lane);
      if (k0 + 32 < Hh)
        __builtin_prefetch(Wh + (size_t)(n0 + (lane & 15)) * Hh + k0 + 32, 0, 3);
      bf16x16 a0 = lds_load_frag(panel, LDP, mtb * 16, KX + k0, lane);
      acc0 = wmma_bf16(a0, bf, acc0);
      bf16x16 a1 = lds_load_frag(panel, LDP, (mtb + 1) * 16, KX + k0, lane);
      acc1 = wmma_bf16(a1, bf, acc1);
    }

    // C/D layout: VGPR r -> M = tile*16 + r + 8*(lane>>4), N = lane&15
#pragma unroll
    for (int r = 0; r < 8; ++r) {
      int m0 = mtb * 16 + r + ((lane >> 4) << 3);
      int n  = lane & 15;
      gbuf[(g * Bsz + m0) * 16 + n]        = acc0[r];
      gbuf[(g * Bsz + m0 + 16) * 16 + n]   = acc1[r];
    }
    __syncthreads();

    // --- cell update ---
#pragma unroll
    for (int e = 0; e < 4; ++e) {
      int p = tid * 4 + e, m = p >> 4, n = p & 15;
      int col  = j * 16 + n;
      float iv = sigm(gbuf[(0 * Bsz + m) * 16 + n] + bI[e]);
      float fv = sigm(gbuf[(1 * Bsz + m) * 16 + n] + bF[e]);
      float gv = tanhf(gbuf[(2 * Bsz + m) * 16 + n] + bG[e]);
      float ov = sigm(gbuf[(3 * Bsz + m) * 16 + n] + bO[e]);
      float c  = fv * cst[e] + iv * gv;
      cst[e]   = c;
      unsigned short hb = f2bf(ov * tanhf(c));
      hnext[m * Hh + col] = hb;
      hout[((size_t)m * Tt + t) * Hh + col] = hb;
    }
    grid_barrier(cnt, gen, NWG);   // includes __syncthreads (LDS reuse safe)
  }
}

// ---------------------------------------------------------------------------
// FC: out[m, n] = h1[m, :] @ Wfc[n, :] + b_fc[n] ;  M=65536, N=128, K=512.
// Block = one 16-row M tile staged once into LDS (shared by all 8 waves),
// 8 waves cover N=128 (one 16x16 tile per wave).
// ---------------------------------------------------------------------------
__global__ __launch_bounds__(256, 1)
void fc_kernel(const unsigned short* __restrict__ h1,
               const unsigned short* __restrict__ Wfc,  // [128, 512]
               const float* __restrict__ bfc,
               float* __restrict__ out) {
  constexpr int LDP = Hh + LDPAD;        // 528
  const int mt   = blockIdx.x;           // 0..4095
  const int tid  = threadIdx.x;
  const int lane = tid & 31;
  const int wave = tid >> 5;
  const int n0   = wave * 16;

  extern __shared__ char smem_raw[];
  lds_us* panel = (lds_us*)smem_raw;     // [16][LDP]

  for (int i = tid; i < 16 * (Hh >> 3); i += 256) {
    int r = i >> 6, c = i & 63;
    cp16_g2l(h1 + ((size_t)(mt * 16 + r)) * Hh + (c << 3),
             panel + r * LDP + (c << 3));
  }
  async_fence();
  __syncthreads();

  v8f acc = {};
  for (int k0 = 0; k0 < Hh; k0 += 32) {
    bf16x16 a = lds_load_frag(panel, LDP, 0, k0, lane);
    bf16x16 b = gload_frag(Wfc, Hh, n0, k0, lane);
    acc = wmma_bf16(a, b, acc);
  }
#pragma unroll
  for (int r = 0; r < 8; ++r) {
    int m = mt * 16 + r + ((lane >> 4) << 3);
    int n = n0 + (lane & 15);
    out[(size_t)m * Oo + n] = acc[r] + bfc[n];
  }
}

// ---------------------------------------------------------------------------
// Host
// ---------------------------------------------------------------------------
extern "C" void kernel_launch(void* const* d_in, const int* in_sizes, int n_in,
                              void* d_out, int out_size, void* d_ws,
                              size_t ws_size, hipStream_t stream) {
  (void)in_sizes; (void)n_in; (void)out_size; (void)ws_size;
  const float* x     = (const float*)d_in[0];
  const float* W_ih0 = (const float*)d_in[1];
  const float* W_hh0 = (const float*)d_in[2];
  const float* b_ih0 = (const float*)d_in[3];
  const float* b_hh0 = (const float*)d_in[4];
  const float* W_ih1 = (const float*)d_in[5];
  const float* W_hh1 = (const float*)d_in[6];
  const float* b_ih1 = (const float*)d_in[7];
  const float* b_hh1 = (const float*)d_in[8];
  const float* W_fc  = (const float*)d_in[9];
  const float* b_fc  = (const float*)d_in[10];
  float* out = (float*)d_out;

  // Workspace carve-up (256B aligned)
  char* w = (char*)d_ws;
  size_t off = 0;
  auto carve = [&](size_t bytes) -> void* {
    void* p = w + off;
    off = (off + bytes + 255) & ~(size_t)255;
    return p;
  };
  int* bar = (int*)carve(2 * sizeof(int));                       // cnt, gen
  unsigned short* x_bf   = (unsigned short*)carve((size_t)Bsz * Tt * Ii * 2);
  unsigned short* h0_bf  = (unsigned short*)carve((size_t)Bsz * Tt * Hh * 2);
  unsigned short* h1_bf  = (unsigned short*)carve((size_t)Bsz * Tt * Hh * 2);
  unsigned short* Wih0_b = (unsigned short*)carve((size_t)G4 * Ii * 2);
  unsigned short* Whh0_b = (unsigned short*)carve((size_t)G4 * Hh * 2);
  unsigned short* Wih1_b = (unsigned short*)carve((size_t)G4 * Hh * 2);
  unsigned short* Whh1_b = (unsigned short*)carve((size_t)G4 * Hh * 2);
  unsigned short* Wfc_b  = (unsigned short*)carve((size_t)Oo * Hh * 2);
  float* bias0 = (float*)carve(G4 * sizeof(float));
  float* bias1 = (float*)carve(G4 * sizeof(float));
  unsigned short* hping  = (unsigned short*)carve((size_t)2 * Bsz * Hh * 2);

  hipLaunchKernelGGL(init_barrier_kernel, dim3(1), dim3(64), 0, stream, bar);

  // f32 -> bf16 conversions
  hipLaunchKernelGGL(cvt_bf16_kernel, dim3(2048), dim3(256), 0, stream,
                     x, x_bf, Bsz * Tt * Ii);
  hipLaunchKernelGGL(cvt_bf16_kernel, dim3(256), dim3(256), 0, stream,
                     W_ih0, Wih0_b, G4 * Ii);
  hipLaunchKernelGGL(cvt_bf16_kernel, dim3(1024), dim3(256), 0, stream,
                     W_hh0, Whh0_b, G4 * Hh);
  hipLaunchKernelGGL(cvt_bf16_kernel, dim3(1024), dim3(256), 0, stream,
                     W_ih1, Wih1_b, G4 * Hh);
  hipLaunchKernelGGL(cvt_bf16_kernel, dim3(1024), dim3(256), 0, stream,
                     W_hh1, Whh1_b, G4 * Hh);
  hipLaunchKernelGGL(cvt_bf16_kernel, dim3(64), dim3(256), 0, stream,
                     W_fc, Wfc_b, Oo * Hh);
  hipLaunchKernelGGL(fuse_bias_kernel, dim3(8), dim3(256), 0, stream,
                     b_ih0, b_hh0, bias0, G4);
  hipLaunchKernelGGL(fuse_bias_kernel, dim3(8), dim3(256), 0, stream,
                     b_ih1, b_hh1, bias1, G4);

  // Dynamic LDS: A-panel + 16KB gate buffer
  const int smem0 = Bsz * (Ii + Hh + LDPAD) * 2 + 4 * Bsz * 16 * 4;  // ~98 KB
  const int smem1 = Bsz * (Hh + Hh + LDPAD) * 2 + 4 * Bsz * 16 * 4;  // ~146 KB
  const int smemF = 16 * (Hh + LDPAD) * 2;                           // ~16.5 KB

  // Layer 0 scan:  x[b,t,i] at b*(T*I) + t*I + i
  hipLaunchKernelGGL((lstm_scan_kernel<Ii>), dim3(NWG), dim3(256), smem0,
                     stream, x_bf, Tt * Ii, Ii, Wih0_b, Whh0_b, bias0,
                     hping, h0_bf, bar + 0, bar + 1);
  // Layer 1 scan:  h0[(b*T+t)*H + k] -> b-stride T*H, t-stride H
  hipLaunchKernelGGL((lstm_scan_kernel<Hh>), dim3(NWG), dim3(256), smem1,
                     stream, h0_bf, Tt * Hh, Hh, Wih1_b, Whh1_b, bias1,
                     hping, h1_bf, bar + 0, bar + 1);
  // FC
  hipLaunchKernelGGL(fc_kernel, dim3((Bsz * Tt) / 16), dim3(256), smemF,
                     stream, h1_bf, Wfc_b, b_fc, out);
}
